// SelfBlock_65137474011239
// MI455X (gfx1250) — compile-verified, matrix operand order checked by reading, not run on hardware
//
#include <hip/hip_runtime.h>
#include <hip/hip_bf16.h>
#include <math.h>

// ---------------------------------------------------------------------------
// Problem constants (match reference)
// ---------------------------------------------------------------------------
enum { Bz = 4, Sq = 1024, Em = 1024, Hh = 16, Dd = 64 };
static const long long BS = (long long)Bz * Sq;          // 4096 rows

typedef __bf16 bf8v  __attribute__((ext_vector_type(8)));
typedef __bf16 bf16v __attribute__((ext_vector_type(16)));
typedef float  f8v   __attribute__((ext_vector_type(8)));
typedef unsigned int u32x4 __attribute__((ext_vector_type(4)));
typedef int          i32x4 __attribute__((ext_vector_type(4)));
typedef int          i32x8 __attribute__((ext_vector_type(8)));

#if __has_builtin(__builtin_amdgcn_tensor_load_to_lds)
#define USE_TDM 1
#else
#define USE_TDM 0
#endif

// ---------------------------------------------------------------------------
// TDM: DMA one 2D bf16 tile (tile_d1 rows x tile_d0 elems, row stride
// stride0 elems) from global memory into LDS at lds_addr, padding each
// 16-DWORD row with 4 DWORDs (-> 40-element LDS row pitch).
// Descriptor packing per CDNA5 ISA ch. 8 (D# groups 0/1).
// ---------------------------------------------------------------------------
#if USE_TDM
__device__ inline void tdm_load_tile_2d(unsigned lds_addr, const void* gptr,
                                        unsigned tile_d0, unsigned tile_d1,
                                        unsigned long long stride0_elems)
{
    const unsigned long long ga = (unsigned long long)gptr;

    u32x4 g0;
    g0[0] = 1u;                                            // count=1, user mode
    g0[1] = lds_addr;                                      // LDS byte address
    g0[2] = (unsigned)(ga & 0xFFFFFFFFu);                  // global_addr[31:0]
    g0[3] = (unsigned)((ga >> 32) & 0x01FFFFFFu)           // global_addr[56:32]
          | 0x80000000u;                                   // type=2 ("image")

    // group1 word0: workgroup_mask=0, data_size=1 (2B), pad_enable=1,
    // pad_interval=3 (16 DWORDs), pad_amount=3 (4 DWORDs)
    const unsigned w0 = (1u << 16) | (1u << 20) | (3u << 22) | (3u << 25);
    const unsigned td0 = 0x7FFFFFFFu;                      // tensor dims: huge
    const unsigned td1 = 0x7FFFFFFFu;                      //  (no OOB clipping)
    const unsigned w1 = (td0 & 0xFFFFu) << 16;             // [47:32]=atomic_bar=0
    const unsigned w2 = ((td0 >> 16) & 0xFFFFu) | ((td1 & 0xFFFFu) << 16);
    const unsigned w3 = ((td1 >> 16) & 0xFFFFu) | (tile_d0 << 16);
    const unsigned w4 = (tile_d1 & 0xFFFFu);               // tile_dim2 = 0
    const unsigned w5 = (unsigned)(stride0_elems & 0xFFFFFFFFull);
    const unsigned w6 = (unsigned)((stride0_elems >> 32) & 0xFFFFull);
    const unsigned w7 = 0u;                                // dim1_stride unused (2D)
    i32x8 g1 = { (int)w0, (int)w1, (int)w2, (int)w3,
                 (int)w4, (int)w5, (int)w6, (int)w7 };
    i32x4 gz = { 0, 0, 0, 0 };
#if __clang_major__ >= 23
    i32x8 gz8 = { 0, 0, 0, 0, 0, 0, 0, 0 };
    __builtin_amdgcn_tensor_load_to_lds(g0, g1, gz, gz, gz8, 0);
#else
    __builtin_amdgcn_tensor_load_to_lds(g0, g1, gz, gz, 0);
#endif
}
#endif

// ---------------------------------------------------------------------------
// Fragment loader: 16-bit A/B operand for v_wmma_f32_16x16x32_bf16.
// Per ISA 7.12.2: lanes 0-15 hold K {0..7, 16..23}; lanes 16-31 hold
// K {8..15, 24..31}. rowPtr points at the lane's (m or n) row in LDS.
// ---------------------------------------------------------------------------
__device__ inline bf16v load_frag16(const __bf16* rowPtr, int lane) {
    const int klo = (lane & 16) ? 8 : 0;
    bf8v lo = *(const bf8v*)(rowPtr + klo);
    bf8v hi = *(const bf8v*)(rowPtr + klo + 16);
    bf16v f;
#pragma unroll
    for (int i = 0; i < 8; ++i) { f[i] = lo[i]; f[i + 8] = hi[i]; }
    return f;
}

// ---------------------------------------------------------------------------
// Generic batched GEMM:  C[z][m][n] = alpha * sum_k A[z][m][k]*W[z][n][k]
//                                   + biasScale*bias[(z/biasDiv)*biasBStride + n]
//                                   + resid[m*N + n]
// A: bf16 [M,K] row-major, W: bf16 [N,K] row-major (i.e. C = A @ W^T).
// M,N multiples of 64; K multiple of 32.
// Block = 128 threads = 4 waves; 64x64 tile; each wave owns 32x32.
// K-blocks are DMA'd into double-buffered LDS by the Tensor Data Mover.
// ---------------------------------------------------------------------------
__global__ __launch_bounds__(128)
void gemm_bf16_wmma(const __bf16* __restrict__ A, const __bf16* __restrict__ W,
                    const float* __restrict__ bias, const float* __restrict__ biasScalePtr,
                    const float* __restrict__ resid, float* __restrict__ C,
                    int M, int N, int K,
                    long long sA, long long sW, long long sC,
                    int biasDiv, long long biasBStride, float alpha)
{
    const int z  = blockIdx.z;
    const int n0 = blockIdx.x * 64;
    const int m0 = blockIdx.y * 64;

    const __bf16* Ab = A + (long long)z * sA;
    const __bf16* Wb = W + (long long)z * sW;
    float*        Cb = C + (long long)z * sC;

    const int tid  = threadIdx.x;
    const int lane = tid & 31;
    const int wid  = tid >> 5;
    const int wr   = wid >> 1;   // wave row quadrant (0..1)
    const int wc   = wid & 1;    // wave col quadrant (0..1)

    __shared__ __align__(16) __bf16 Ash[2][64][40];  // 32 K + 8 pad (80B rows)
    __shared__ __align__(16) __bf16 Wsh[2][64][40];

    f8v acc[2][2];
#pragma unroll
    for (int i = 0; i < 2; ++i)
#pragma unroll
        for (int j = 0; j < 2; ++j) acc[i][j] = (f8v)0.0f;

    const __bf16* Atile = Ab + (long long)m0 * K;    // 64 rows of A
    const __bf16* Wtile = Wb + (long long)n0 * K;    // 64 rows of W

#if USE_TDM
    const unsigned ldsA[2] = { (unsigned)(unsigned long long)(void*)&Ash[0][0][0],
                               (unsigned)(unsigned long long)(void*)&Ash[1][0][0] };
    const unsigned ldsW[2] = { (unsigned)(unsigned long long)(void*)&Wsh[0][0][0],
                               (unsigned)(unsigned long long)(void*)&Wsh[1][0][0] };
    if (wid == 0) {
        tdm_load_tile_2d(ldsA[0], Atile, 32, 64, (unsigned long long)K);
        tdm_load_tile_2d(ldsW[0], Wtile, 32, 64, (unsigned long long)K);
        __builtin_amdgcn_s_wait_tensorcnt(0);
    }
    __syncthreads();
#else
    const int lr = tid >> 1;          // 0..63 : tile row this thread stages
    const int lc = (tid & 1) * 16;    // 0 or 16: K-chunk this thread stages
    {
        bf8v a0 = ((const bf8v*)(Atile + (long long)lr * K + lc))[0];
        bf8v a1 = ((const bf8v*)(Atile + (long long)lr * K + lc))[1];
        bf8v w0 = ((const bf8v*)(Wtile + (long long)lr * K + lc))[0];
        bf8v w1 = ((const bf8v*)(Wtile + (long long)lr * K + lc))[1];
        *(bf8v*)&Ash[0][lr][lc]     = a0;
        *(bf8v*)&Ash[0][lr][lc + 8] = a1;
        *(bf8v*)&Wsh[0][lr][lc]     = w0;
        *(bf8v*)&Wsh[0][lr][lc + 8] = w1;
    }
    __syncthreads();
#endif

    int cur = 0;
    for (int kb = 0; kb < K; kb += 32) {
        const int nxt = cur ^ 1;
        const bool more = (kb + 32) < K;

#if USE_TDM
        if (more && wid == 0) {       // DMA next K-block while we compute
            tdm_load_tile_2d(ldsA[nxt], Atile + kb + 32, 32, 64, (unsigned long long)K);
            tdm_load_tile_2d(ldsW[nxt], Wtile + kb + 32, 32, 64, (unsigned long long)K);
        }
#else
        bf8v a0, a1, w0, w1;
        if (more) {
            a0 = ((const bf8v*)(Atile + (long long)lr * K + kb + 32 + lc))[0];
            a1 = ((const bf8v*)(Atile + (long long)lr * K + kb + 32 + lc))[1];
            w0 = ((const bf8v*)(Wtile + (long long)lr * K + kb + 32 + lc))[0];
            w1 = ((const bf8v*)(Wtile + (long long)lr * K + kb + 32 + lc))[1];
        }
#endif

        bf16v fa[2], fb[2];
#pragma unroll
        for (int ti = 0; ti < 2; ++ti)
            fa[ti] = load_frag16(&Ash[cur][32 * wr + 16 * ti + (lane & 15)][0], lane);
#pragma unroll
        for (int tj = 0; tj < 2; ++tj)
            fb[tj] = load_frag16(&Wsh[cur][32 * wc + 16 * tj + (lane & 15)][0], lane);

#pragma unroll
        for (int ti = 0; ti < 2; ++ti)
#pragma unroll
            for (int tj = 0; tj < 2; ++tj)
                acc[ti][tj] = __builtin_amdgcn_wmma_f32_16x16x32_bf16(
                    false, fa[ti], false, fb[tj], (short)0, acc[ti][tj],
                    false, false);

#if USE_TDM
        if (more && wid == 0) __builtin_amdgcn_s_wait_tensorcnt(0);
#else
        if (more) {
            *(bf8v*)&Ash[nxt][lr][lc]     = a0;
            *(bf8v*)&Ash[nxt][lr][lc + 8] = a1;
            *(bf8v*)&Wsh[nxt][lr][lc]     = w0;
            *(bf8v*)&Wsh[nxt][lr][lc + 8] = w1;
        }
#endif
        __syncthreads();
        cur = nxt;
    }

    // ---- epilogue: scale, bias (optionally per-batch & device-scaled), residual
    const float bscale = biasScalePtr ? biasScalePtr[0] : 1.0f;
    const long long bOff = (biasDiv > 0) ? (long long)(z / biasDiv) * biasBStride : 0;

#pragma unroll
    for (int ti = 0; ti < 2; ++ti) {
        const int mbase = m0 + 32 * wr + 16 * ti + ((lane & 16) ? 8 : 0);
#pragma unroll
        for (int tj = 0; tj < 2; ++tj) {
            const int n = n0 + 32 * wc + 16 * tj + (lane & 15);
            const float bv = bias ? bias[bOff + n] * bscale : 0.0f;
#pragma unroll
            for (int i = 0; i < 8; ++i) {
                const int m = mbase + i;
                float v = acc[ti][tj][i] * alpha + bv;
                if (resid) v += resid[(long long)m * N + n];
                Cb[(long long)m * N + n] = v;
            }
        }
    }
}

// ---------------------------------------------------------------------------
// Elementwise fp32 -> bf16 cast
// ---------------------------------------------------------------------------
__global__ void cast_f32_bf16(const float* __restrict__ in, __bf16* __restrict__ out,
                              long long n)
{
    long long i = (long long)blockIdx.x * blockDim.x + threadIdx.x;
    if (i < n) out[i] = (__bf16)in[i];
}

// ---------------------------------------------------------------------------
// RoPE + interleaved QKV split.
// qkv: [B,S,3E] f32, channel c = h*192 + d*3 + comp.
// Outputs: q,k bf16 [B,H,S,D] (rotary applied), vT bf16 [B,H,D,S].
// ---------------------------------------------------------------------------
__global__ void rope_split(const float* __restrict__ qkv, const float* __restrict__ enc,
                           __bf16* __restrict__ q, __bf16* __restrict__ k,
                           __bf16* __restrict__ vT)
{
    const long long total = (long long)Bz * Hh * Sq * (Dd / 2);
    long long idx = (long long)blockIdx.x * blockDim.x + threadIdx.x;
    if (idx >= total) return;

    const int  p  = (int)(idx & (Dd / 2 - 1));
    long long  r  = idx >> 5;              // /(D/2)
    const int  s  = (int)(r & (Sq - 1));
    const long long bh = r >> 10;          // b*H + h
    const int  h  = (int)(bh & (Hh - 1));
    const int  b  = (int)(bh >> 4);

    const int d0 = 2 * p, d1 = d0 + 1;
    const float* base = qkv + ((long long)b * Sq + s) * (3 * Em) + h * (Dd * 3);
    const float q0 = base[d0 * 3 + 0], q1 = base[d1 * 3 + 0];
    const float k0 = base[d0 * 3 + 1], k1 = base[d1 * 3 + 1];
    const float v0 = base[d0 * 3 + 2], v1 = base[d1 * 3 + 2];

    const float* cs = enc + (long long)s * Dd;                      // encoding[0]
    const float* sn = enc + (long long)Sq * Dd + (long long)s * Dd; // encoding[1]
    const float c0 = cs[d0], c1 = cs[d1], s0 = sn[d0], s1 = sn[d1];

    const float qr0 = q0 * c0 - q1 * s0;
    const float qr1 = q1 * c1 + q0 * s1;
    const float kr0 = k0 * c0 - k1 * s0;
    const float kr1 = k1 * c1 + k0 * s1;

    const long long o = (bh * Sq + s) * Dd;
    q[o + d0] = (__bf16)qr0;  q[o + d1] = (__bf16)qr1;
    k[o + d0] = (__bf16)kr0;  k[o + d1] = (__bf16)kr1;
    vT[(bh * Dd + d0) * Sq + s] = (__bf16)v0;
    vT[(bh * Dd + d1) * Sq + s] = (__bf16)v1;
}

// ---------------------------------------------------------------------------
// In-place row softmax over S=1024 logits; also emits bf16 copy for attn@v.
// ---------------------------------------------------------------------------
__global__ __launch_bounds__(256)
void softmax_rows(float* __restrict__ attn, __bf16* __restrict__ attn_bf)
{
    const long long row = blockIdx.x;
    float*  p  = attn    + row * (long long)Sq;
    __bf16* pb = attn_bf + row * (long long)Sq;
    __shared__ float red[256];
    const int t = threadIdx.x;

    float lm = -INFINITY;
    for (int j = t; j < Sq; j += 256) lm = fmaxf(lm, p[j]);
    red[t] = lm; __syncthreads();
    for (int s = 128; s > 0; s >>= 1) { if (t < s) red[t] = fmaxf(red[t], red[t + s]); __syncthreads(); }
    const float mx = red[0]; __syncthreads();

    float ls = 0.0f;
    for (int j = t; j < Sq; j += 256) ls += expf(p[j] - mx);
    red[t] = ls; __syncthreads();
    for (int s = 128; s > 0; s >>= 1) { if (t < s) red[t] += red[t + s]; __syncthreads(); }
    const float inv = 1.0f / red[0];

    for (int j = t; j < Sq; j += 256) {
        const float v = expf(p[j] - mx) * inv;
        p[j]  = v;
        pb[j] = (__bf16)v;
    }
}

// ctx [B,H,S,D] f32 -> msg-input bf16 [B,S,E] (head-major -> embed-major)
__global__ void ctx_transpose(const float* __restrict__ ctx, __bf16* __restrict__ out)
{
    const long long total = (long long)Bz * Sq * Em;
    long long idx = (long long)blockIdx.x * blockDim.x + threadIdx.x;
    if (idx >= total) return;
    const int e = (int)(idx & (Em - 1));
    const int s = (int)((idx >> 10) & (Sq - 1));
    const int b = (int)(idx >> 20);
    const int h = e >> 6, d = e & 63;
    out[idx] = (__bf16)ctx[(((long long)b * Hh + h) * Sq + s) * Dd + d];
}

// hcat bf16 [BS, 2E] = concat(x, message)
__global__ void build_hcat(const float* __restrict__ x, const float* __restrict__ msg,
                           __bf16* __restrict__ hcat)
{
    const long long total = BS * (long long)(2 * Em);
    long long idx = (long long)blockIdx.x * blockDim.x + threadIdx.x;
    if (idx >= total) return;
    const long long m = idx >> 11;          // /2048
    const int c = (int)(idx & 2047);
    const float v = (c < Em) ? x[m * Em + c] : msg[m * Em + (c - Em)];
    hcat[idx] = (__bf16)v;
}

// LayerNorm over 2E=2048 + exact GELU, bf16 out.
__global__ __launch_bounds__(256)
void ln_gelu(const float* __restrict__ h, const float* __restrict__ g,
             const float* __restrict__ bta, __bf16* __restrict__ out)
{
    const int Cc = 2 * Em;
    const long long row = blockIdx.x;
    const float* p = h + row * (long long)Cc;
    __shared__ float red[256];
    const int t = threadIdx.x;

    float s1 = 0.0f, s2 = 0.0f;
    for (int j = t; j < Cc; j += 256) { const float v = p[j]; s1 += v; s2 += v * v; }
    red[t] = s1; __syncthreads();
    for (int s = 128; s > 0; s >>= 1) { if (t < s) red[t] += red[t + s]; __syncthreads(); }
    const float mu = red[0] / Cc; __syncthreads();
    red[t] = s2; __syncthreads();
    for (int s = 128; s > 0; s >>= 1) { if (t < s) red[t] += red[t + s]; __syncthreads(); }
    const float var = red[0] / Cc - mu * mu;
    const float rs  = rsqrtf(var + 1e-5f);

    for (int j = t; j < Cc; j += 256) {
        const float y  = (p[j] - mu) * rs * g[j] + bta[j];
        const float ge = 0.5f * y * (1.0f + erff(y * 0.70710678118654752f));
        out[row * (long long)Cc + j] = (__bf16)ge;
    }
}

// ---------------------------------------------------------------------------
// Host-side launch
// ---------------------------------------------------------------------------
static inline void launch_gemm(hipStream_t st, const __bf16* A, const __bf16* W,
                               const float* bias, const float* bscale,
                               const float* resid, float* C,
                               int M, int N, int K,
                               long long sA, long long sW, long long sC, int Z,
                               int biasDiv, long long biasBStride, float alpha)
{
    dim3 grid(N / 64, M / 64, Z);
    gemm_bf16_wmma<<<grid, 128, 0, st>>>(A, W, bias, bscale, resid, C,
                                         M, N, K, sA, sW, sC,
                                         biasDiv, biasBStride, alpha);
}

static inline void launch_cast(hipStream_t st, const float* in, __bf16* out, long long n)
{
    cast_f32_bf16<<<(unsigned)((n + 255) / 256), 256, 0, st>>>(in, out, n);
}

extern "C" void kernel_launch(void* const* d_in, const int* in_sizes, int n_in,
                              void* d_out, int out_size, void* d_ws, size_t ws_size,
                              hipStream_t stream)
{
    (void)in_sizes; (void)n_in; (void)out_size; (void)ws_size;

    const float* x       = (const float*)d_in[0];
    const float* enc     = (const float*)d_in[1];
    const float* wmask   = (const float*)d_in[2];
    const float* Wqkv_w  = (const float*)d_in[3];
    const float* Wqkv_b  = (const float*)d_in[4];
    const float* wscale  = (const float*)d_in[5];
    const float* out_w   = (const float*)d_in[6];
    const float* out_b   = (const float*)d_in[7];
    const float* ffn1_w  = (const float*)d_in[8];
    const float* ffn1_b  = (const float*)d_in[9];
    const float* ln_g    = (const float*)d_in[10];
    const float* ln_b    = (const float*)d_in[11];
    const float* ffn2_w  = (const float*)d_in[12];
    const float* ffn2_b  = (const float*)d_in[13];

    float* out_f32  = (float*)d_out;                       // [B,S,E]
    float* attn_f32 = out_f32 + BS * (long long)Em;        // [B,H,S,S]

    // workspace carve-up (256B aligned)
    char*  base = (char*)d_ws;
    size_t off  = 0;
    auto alloc = [&](size_t bytes) -> void* {
        void* p = base + off;
        off = (off + bytes + 255) & ~(size_t)255;
        return p;
    };
    __bf16* x_bf     = (__bf16*)alloc(BS * Em * 2);
    __bf16* wqkv_bf  = (__bf16*)alloc((size_t)3 * Em * Em * 2);
    __bf16* outw_bf  = (__bf16*)alloc((size_t)Em * Em * 2);
    __bf16* ffn1_bf  = (__bf16*)alloc((size_t)2 * Em * 2 * Em * 2);
    __bf16* ffn2_bf  = (__bf16*)alloc((size_t)Em * 2 * Em * 2);
    float*  qkv_f    = (float*) alloc(BS * 3 * Em * 4);
    __bf16* q_bf     = (__bf16*)alloc((size_t)Bz * Hh * Sq * Dd * 2);
    __bf16* k_bf     = (__bf16*)alloc((size_t)Bz * Hh * Sq * Dd * 2);
    __bf16* vT_bf    = (__bf16*)alloc((size_t)Bz * Hh * Dd * Sq * 2);
    __bf16* attn_bf  = (__bf16*)alloc((size_t)Bz * Hh * Sq * Sq * 2);
    float*  ctx_f    = (float*) alloc((size_t)Bz * Hh * Sq * Dd * 4);
    __bf16* msgin_bf = (__bf16*)alloc(BS * Em * 2);
    float*  msg_f    = (float*) alloc(BS * Em * 4);
    __bf16* hcat_bf  = (__bf16*)alloc(BS * 2 * Em * 2);
    float*  h_f      = (float*) alloc(BS * 2 * Em * 4);
    __bf16* hact_bf  = (__bf16*)alloc(BS * 2 * Em * 2);

    // 1) fp32 -> bf16 operand casts
    launch_cast(stream, x,      x_bf,    BS * Em);
    launch_cast(stream, Wqkv_w, wqkv_bf, (long long)3 * Em * Em);
    launch_cast(stream, out_w,  outw_bf, (long long)Em * Em);
    launch_cast(stream, ffn1_w, ffn1_bf, (long long)2 * Em * 2 * Em);
    launch_cast(stream, ffn2_w, ffn2_bf, (long long)Em * 2 * Em);

    // 2) QKV projection: [4096,3072] = x @ Wqkv^T + b
    launch_gemm(stream, x_bf, wqkv_bf, Wqkv_b, nullptr, nullptr, qkv_f,
                (int)BS, 3 * Em, Em, 0, 0, 0, 1, 0, 0, 1.0f);

    // 3) RoPE + interleaved split -> q,k bf16 [B,H,S,D], vT bf16 [B,H,D,S]
    {
        long long tot = (long long)Bz * Hh * Sq * (Dd / 2);
        rope_split<<<(unsigned)((tot + 255) / 256), 256, 0, stream>>>(qkv_f, enc, q_bf, k_bf, vT_bf);
    }

    // 4) logits = q @ k^T * D^-0.5 + wmask*wscale  -> straight into d_out attn area
    launch_gemm(stream, q_bf, k_bf, wmask, wscale, nullptr, attn_f32,
                Sq, Sq, Dd,
                (long long)Sq * Dd, (long long)Sq * Dd, (long long)Sq * Sq,
                Bz * Hh, Hh, Sq, 0.125f);

    // 5) softmax in place (+ bf16 copy)
    softmax_rows<<<(unsigned)(Bz * Hh * Sq), 256, 0, stream>>>(attn_f32, attn_bf);

    // 6) ctx = attn @ v  (W operand = vT [D,S])
    launch_gemm(stream, attn_bf, vT_bf, nullptr, nullptr, nullptr, ctx_f,
                Sq, Dd, Sq,
                (long long)Sq * Sq, (long long)Dd * Sq, (long long)Sq * Dd,
                Bz * Hh, 0, 0, 1.0f);

    // 7) head-major -> embed-major bf16
    {
        long long tot = BS * Em;
        ctx_transpose<<<(unsigned)((tot + 255) / 256), 256, 0, stream>>>(ctx_f, msgin_bf);
    }

    // 8) message = ctx @ out_w^T + out_b
    launch_gemm(stream, msgin_bf, outw_bf, out_b, nullptr, nullptr, msg_f,
                (int)BS, Em, Em, 0, 0, 0, 1, 0, 0, 1.0f);

    // 9) hcat = concat(x, message) in bf16
    {
        long long tot = BS * 2 * Em;
        build_hcat<<<(unsigned)((tot + 255) / 256), 256, 0, stream>>>(x, msg_f, hcat_bf);
    }

    // 10) FFN1: [4096,2048] = hcat @ ffn1^T + b
    launch_gemm(stream, hcat_bf, ffn1_bf, ffn1_b, nullptr, nullptr, h_f,
                (int)BS, 2 * Em, 2 * Em, 0, 0, 0, 1, 0, 0, 1.0f);

    // 11) LayerNorm + exact GELU -> bf16
    ln_gelu<<<(unsigned)BS, 256, 0, stream>>>(h_f, ln_g, ln_b, hact_bf);

    // 12) FFN2 + residual: out = x + hact @ ffn2^T + b  -> d_out
    launch_gemm(stream, hact_bf, ffn2_bf, ffn2_b, nullptr, x, out_f32,
                (int)BS, Em, 2 * Em, 0, 0, 0, 1, 0, 0, 1.0f);
}
